// CrossAttnDelta_18837726560323
// MI455X (gfx1250) — matrix operand order repaired and checked by reading, not compiled, not run on hardware
//
#include <hip/hip_runtime.h>

#define C_DIM 512    // AUDIO_DIM
#define D_DIM 768    // SEM_DIM
#define A_DIM 128    // ATTN_DIM
#define BATCH 8
#define TLEN  4096
#define SLEN  1024

typedef __attribute__((ext_vector_type(16))) _Float16 v16h;
typedef __attribute__((ext_vector_type(8)))  _Float16 v8h;
typedef __attribute__((ext_vector_type(8)))  float    v8f;

__device__ __forceinline__ v8f wmma32(v16h a, v16h b, v8f c) {
  // D(f32 16x16) = A(f16 16x32) * B(f16 32x16) + C
  return __builtin_amdgcn_wmma_f32_16x16x32_f16(
      /*neg_a=*/false, a, /*neg_b=*/false, b,
      /*c_mod=*/(short)0, c, /*reuse_a=*/false, /*reuse_b=*/false);
}

// A-fragment K packing: element e at lane-half hi holds k = (e&7)+8*hi+16*(e>>3)
__device__ __forceinline__ int kofs(int e, int hi) {
  return (e & 7) + (hi << 3) + ((e >> 3) << 4);
}

__device__ __forceinline__ v16h cat8h(v8h a, v8h b) {
  return __builtin_shufflevector(a, b, 0, 1, 2, 3, 4, 5, 6, 7,
                                 8, 9, 10, 11, 12, 13, 14, 15);
}

// Convert two contiguous 8-float runs into a packed f16 fragment half-pair.
__device__ __forceinline__ v16h cvt16(v8f lo, v8f hi) {
  v16h r;
#pragma unroll
  for (int e = 0; e < 8; ++e) {
    r[e]     = (_Float16)lo[e];
    r[e + 8] = (_Float16)hi[e];
  }
  return r;
}

// Load a B fragment from a row-major f32 matrix column (lane = col, elem = k via kofs).
__device__ __forceinline__ v16h load_bfrag_f32row(const float* col, int ld, int hi) {
  v16h r;
#pragma unroll
  for (int e = 0; e < 16; ++e)
    r[e] = (_Float16)col[(size_t)kofs(e, hi) * ld];
  return r;
}

// ---------------------------------------------------------------------------
// Kernel 1: LayerNorm over channels + Q projection.
// One wave handles a 16-row tile of T. Grid = B*T/16 tiles, 8 waves/block.
// ---------------------------------------------------------------------------
__global__ __launch_bounds__(256) void ln_q_proj(
    const float* __restrict__ x,   // [B, C, T]
    const float* __restrict__ lnw, const float* __restrict__ lnb,
    const float* __restrict__ Wq,  // [C, A]
    const float* __restrict__ bq,
    _Float16* __restrict__ Qh)     // [B, T, A] f16
{
  const int lane = threadIdx.x & 31;
  const int wave = threadIdx.x >> 5;
  const int tile = blockIdx.x * 8 + wave;
  const int b    = tile / (TLEN / 16);
  const int t0   = (tile % (TLEN / 16)) * 16;
  const int ln15 = lane & 15;
  const int hi   = lane >> 4;
  const int t    = t0 + ln15;
  const size_t xbase = (size_t)b * C_DIM * TLEN + t;

  // LayerNorm stats for row t: lane pair (l, l+16) splits the 512 channels.
  float sum = 0.f, sq = 0.f;
  for (int c = hi; c < C_DIM; c += 2) {
    float v = x[xbase + (size_t)c * TLEN];
    __builtin_prefetch(&x[xbase + (size_t)(c + 32) * TLEN], 0, 1);
    sum += v;
    sq  += v * v;
  }
  sum += __shfl_xor(sum, 16, 32);
  sq  += __shfl_xor(sq, 16, 32);
  const float mean = sum * (1.f / C_DIM);
  const float rstd = rsqrtf(sq * (1.f / C_DIM) - mean * mean + 1e-5f);

  v8f acc[8];
#pragma unroll
  for (int nt = 0; nt < 8; ++nt) acc[nt] = {};

  for (int kt = 0; kt < C_DIM / 32; ++kt) {
    // A fragment: normalized x row (lane = row, packed channels). x is
    // channel-strided so these stay scalar loads (coalesced across lanes).
    v16h af;
#pragma unroll
    for (int e = 0; e < 16; ++e) {
      const int c = kt * 32 + kofs(e, hi);
      const float v = x[xbase + (size_t)c * TLEN];
      af[e] = (_Float16)((v - mean) * rstd * lnw[c] + lnb[c]);
    }
    // B fragments: lane = output col (contiguous), element = k row of Wq via kofs
    const float* wrow = Wq + (size_t)(kt * 32) * A_DIM + ln15;
#pragma unroll
    for (int nt = 0; nt < 8; ++nt)
      acc[nt] = wmma32(af, load_bfrag_f32row(wrow + nt * 16, A_DIM, hi), acc[nt]);
  }

#pragma unroll
  for (int nt = 0; nt < 8; ++nt) {
#pragma unroll
    for (int r = 0; r < 8; ++r) {
      const int m   = r + hi * 8;
      const int col = nt * 16 + ln15;
      Qh[((size_t)b * TLEN + t0 + m) * A_DIM + col] =
          (_Float16)(acc[nt][r] + bq[col]);
    }
  }
}

// ---------------------------------------------------------------------------
// Kernel 2: K (stored transposed [B,A,S]) and V ([B,S,A]) projections.
// One wave = 16-row tile of S.
// ---------------------------------------------------------------------------
__global__ __launch_bounds__(256) void kv_proj(
    const float* __restrict__ sem,  // [B, S, D]
    const float* __restrict__ Wk, const float* __restrict__ bk,
    const float* __restrict__ Wv, const float* __restrict__ bv,
    _Float16* __restrict__ KTh,     // [B, A, S] f16 (transposed)
    _Float16* __restrict__ Vh)      // [B, S, A] f16
{
  const int lane = threadIdx.x & 31;
  const int wave = threadIdx.x >> 5;
  const int tile = blockIdx.x * 8 + wave;
  const int b    = tile / (SLEN / 16);
  const int s0   = (tile % (SLEN / 16)) * 16;
  const int ln15 = lane & 15;
  const int hi   = lane >> 4;
  const float* arow = sem + ((size_t)b * SLEN + s0 + ln15) * D_DIM;

  for (int pass = 0; pass < 2; ++pass) {
    const float* W    = pass ? Wv : Wk;
    const float* bias = pass ? bv : bk;

    v8f acc[8];
#pragma unroll
    for (int nt = 0; nt < 8; ++nt) acc[nt] = {};

    for (int kt = 0; kt < D_DIM / 32; ++kt) {
      // A fragment: contiguous sem row -> two v8f vector loads
      v8f a0 = *(const v8f*)(arow + kt * 32 + hi * 8);
      v8f a1 = *(const v8f*)(arow + kt * 32 + 16 + hi * 8);
      v16h af = cvt16(a0, a1);
      const float* wrow = W + (size_t)(kt * 32) * A_DIM + ln15;
#pragma unroll
      for (int nt = 0; nt < 8; ++nt)
        acc[nt] = wmma32(af, load_bfrag_f32row(wrow + nt * 16, A_DIM, hi), acc[nt]);
    }
#pragma unroll
    for (int nt = 0; nt < 8; ++nt) {
#pragma unroll
      for (int r = 0; r < 8; ++r) {
        const int m   = s0 + r + hi * 8;         // sem position s
        const int col = nt * 16 + ln15;          // attn dim a
        const _Float16 v = (_Float16)(acc[nt][r] + bias[col]);
        if (pass == 0)
          KTh[((size_t)b * A_DIM + col) * SLEN + m] = v;   // transposed
        else
          Vh[((size_t)b * SLEN + m) * A_DIM + col] = v;    // row-major
      }
    }
  }
}

// ---------------------------------------------------------------------------
// Kernel 3: attention (QK^T -> softmax in LDS -> PV) + output projection,
// writing the delta transposed back to [B, C, T].
// One block (4 waves, 128 threads) per (b, 16-row T tile).
// Dynamic LDS: 16*1024 f32 scores, later reused as 16*512 f32 delta staging.
// ---------------------------------------------------------------------------
__global__ __launch_bounds__(128) void attn_out(
    const _Float16* __restrict__ Qh,   // [B, T, A]
    const _Float16* __restrict__ KTh,  // [B, A, S]
    const _Float16* __restrict__ Vh,   // [B, S, A]
    const float* __restrict__ Wo,      // [A, C]
    const float* __restrict__ bo,
    float* __restrict__ out)           // [B, C, T]
{
  extern __shared__ char smem[];
  float* sc = (float*)smem;                 // [16][1024] then [16][512]
  __shared__ _Float16 ctxh[16 * A_DIM];     // 4 KB
  __shared__ float red[16][9];
  __shared__ float rowstat[16];

  const int tid  = threadIdx.x;
  const int lane = tid & 31;
  const int wave = tid >> 5;
  const int ln15 = lane & 15;
  const int hi   = lane >> 4;
  const int b    = blockIdx.x / (TLEN / 16);
  const int t0   = (blockIdx.x % (TLEN / 16)) * 16;
  const float scale = 0.08838834764831845f;  // 1/sqrt(128)

  // Q A-fragments for the whole S loop: two contiguous 8-half runs each.
  v16h qa[4];
  const _Float16* qrow = Qh + ((size_t)b * TLEN + t0 + ln15) * A_DIM;
#pragma unroll
  for (int kt = 0; kt < 4; ++kt) {
    v8h lo = *(const v8h*)(qrow + kt * 32 + hi * 8);
    v8h up = *(const v8h*)(qrow + kt * 32 + 16 + hi * 8);
    qa[kt] = cat8h(lo, up);
  }

  // ---- scores: each wave covers S tiles st = wave, wave+4, ... ----
  for (int st = wave; st < SLEN / 16; st += 4) {
    const int s0 = st * 16;
    v8f a = {};
#pragma unroll
    for (int kt = 0; kt < 4; ++kt) {
      // B fragment = K^T column (lane = s, elem = attn-dim via kofs): strided gather
      const _Float16* kcol = KTh + ((size_t)b * A_DIM + kt * 32) * SLEN + s0 + ln15;
      v16h kf;
#pragma unroll
      for (int e = 0; e < 16; ++e) kf[e] = kcol[(size_t)kofs(e, hi) * SLEN];
      a = wmma32(qa[kt], kf, a);
    }
#pragma unroll
    for (int r = 0; r < 8; ++r)
      sc[(r + hi * 8) * SLEN + s0 + ln15] = a[r] * scale;
  }
  __syncthreads();

  // ---- softmax over LDS rows: 8 threads per row ----
  {
    const int row = tid >> 3;
    const int j   = tid & 7;
    float mx = -3.4e38f;
    for (int s = j; s < SLEN; s += 8) mx = fmaxf(mx, sc[row * SLEN + s]);
    red[row][j] = mx;
    __syncthreads();
    if (j == 0) {
      float m = red[row][0];
#pragma unroll
      for (int q = 1; q < 8; ++q) m = fmaxf(m, red[row][q]);
      rowstat[row] = m;
    }
    __syncthreads();
    const float rm = rowstat[row];
    float sm = 0.f;
    for (int s = j; s < SLEN; s += 8) {
      const float e = __expf(sc[row * SLEN + s] - rm);
      sc[row * SLEN + s] = e;
      sm += e;
    }
    red[row][j] = sm;
    __syncthreads();
    if (j == 0) {
      float tsum = red[row][0];
#pragma unroll
      for (int q = 1; q < 8; ++q) tsum += red[row][q];
      rowstat[row] = 1.f / tsum;
    }
    __syncthreads();
    const float inv = rowstat[row];
    for (int s = j; s < SLEN; s += 8) sc[row * SLEN + s] *= inv;
  }
  __syncthreads();

  // ---- ctx = P @ V : each wave owns 2 adjacent attn-dim col tiles; the P
  // fragment is built once per s-chunk and reused for both WMMAs. ----
  {
    v8f accv0 = {}, accv1 = {};
    for (int sk = 0; sk < SLEN / 32; ++sk) {
      const int skb = sk * 32;
      v8f p0 = *(const v8f*)(&sc[ln15 * SLEN + skb + hi * 8]);
      v8f p1 = *(const v8f*)(&sc[ln15 * SLEN + skb + 16 + hi * 8]);
      v16h pf = cvt16(p0, p1);
      const _Float16* vrow =
          Vh + ((size_t)b * SLEN + skb) * A_DIM + wave * 32 + ln15;
      v16h vf0, vf1;
#pragma unroll
      for (int e = 0; e < 16; ++e) {
        vf0[e] = vrow[kofs(e, hi) * A_DIM];
        vf1[e] = vrow[kofs(e, hi) * A_DIM + 16];
      }
      accv0 = wmma32(pf, vf0, accv0);
      accv1 = wmma32(pf, vf1, accv1);
    }
#pragma unroll
    for (int r = 0; r < 8; ++r) {
      ctxh[(r + hi * 8) * A_DIM + wave * 32 + ln15]      = (_Float16)accv0[r];
      ctxh[(r + hi * 8) * A_DIM + wave * 32 + 16 + ln15] = (_Float16)accv1[r];
    }
  }
  __syncthreads();

  // ---- delta = ctx @ Wo : 32 col tiles over C=512, 8 per wave ----
  v16h ca[4];
#pragma unroll
  for (int kt = 0; kt < 4; ++kt) {
    v8h lo = *(const v8h*)(&ctxh[ln15 * A_DIM + kt * 32 + hi * 8]);
    v8h up = *(const v8h*)(&ctxh[ln15 * A_DIM + kt * 32 + 16 + hi * 8]);
    ca[kt] = cat8h(lo, up);
  }

  for (int ct = wave; ct < C_DIM / 16; ct += 4) {
    const int c0 = ct * 16;
    v8f a = {};
#pragma unroll
    for (int kt = 0; kt < 4; ++kt) {
      const float* wrow = Wo + (size_t)(kt * 32) * C_DIM + c0 + ln15;
      a = wmma32(ca[kt], load_bfrag_f32row(wrow, C_DIM, hi), a);
    }
#pragma unroll
    for (int r = 0; r < 8; ++r)
      sc[(r + hi * 8) * C_DIM + c0 + ln15] = a[r];  // reuse score LDS
  }
  __syncthreads();

  // ---- coalesced transposed writeback: out[b][c][t0+tt] ----
  for (int idx = tid; idx < C_DIM * 16; idx += 128) {
    const int c  = idx >> 4;
    const int tt = idx & 15;
    out[((size_t)b * C_DIM + c) * TLEN + t0 + tt] = sc[tt * C_DIM + c] + bo[c];
  }
}

// ---------------------------------------------------------------------------
extern "C" void kernel_launch(void* const* d_in, const int* in_sizes, int n_in,
                              void* d_out, int out_size, void* d_ws, size_t ws_size,
                              hipStream_t stream) {
  const float* x   = (const float*)d_in[0];
  const float* sem = (const float*)d_in[1];
  const float* lnw = (const float*)d_in[2];
  const float* lnb = (const float*)d_in[3];
  const float* Wq  = (const float*)d_in[4];
  const float* bq  = (const float*)d_in[5];
  const float* Wk  = (const float*)d_in[6];
  const float* bk  = (const float*)d_in[7];
  const float* Wv  = (const float*)d_in[8];
  const float* bv  = (const float*)d_in[9];
  const float* Wo  = (const float*)d_in[10];
  const float* bo  = (const float*)d_in[11];
  float* out = (float*)d_out;

  _Float16* Qh  = (_Float16*)d_ws;                        // 8 MB
  _Float16* KTh = Qh + (size_t)BATCH * TLEN * A_DIM;      // 2 MB
  _Float16* Vh  = KTh + (size_t)BATCH * A_DIM * SLEN;     // 2 MB

  ln_q_proj<<<BATCH * TLEN / 128, 256, 0, stream>>>(x, lnw, lnb, Wq, bq, Qh);
  kv_proj<<<BATCH * SLEN / 128, 256, 0, stream>>>(sem, Wk, bk, Wv, bv, KTh, Vh);
  attn_out<<<BATCH * TLEN / 16, 128, 16 * SLEN * (int)sizeof(float), stream>>>(
      Qh, KTh, Vh, Wo, bo, out);
}
